// CompGCNTransE_944892805205
// MI455X (gfx1250) — matrix operand order repaired and checked by reading, not compiled, not run on hardware
//
#include <hip/hip_runtime.h>
#include <hip/hip_bf16.h>

#define DIM 128
#define GAMMA 12.0f
#define BN_EPS 1e-5f

typedef __attribute__((ext_vector_type(16))) __bf16 v16bf;
typedef __attribute__((ext_vector_type(8)))  float  v8f;

// Native (unsafe) f32 atomic add -> global_atomic_add_f32, never a CAS loop.
__device__ __forceinline__ void atomic_add_f32(float* p, float v) {
    unsafeAtomicAdd(p, v);
}

// ---------------------------------------------------------------------------
// Generic zero kernel (grid-stride)
// ---------------------------------------------------------------------------
__global__ void zero_kernel(float* __restrict__ p, size_t n) {
    size_t i = (size_t)blockIdx.x * blockDim.x + threadIdx.x;
    size_t stride = (size_t)gridDim.x * blockDim.x;
    for (; i < n; i += stride) p[i] = 0.0f;
}

// ---------------------------------------------------------------------------
// Degree counts (same for both layers): deg_o[dst]++ , deg_i[src]++
// ---------------------------------------------------------------------------
__global__ void degree_kernel(const int* __restrict__ src, const int* __restrict__ dst,
                              int* __restrict__ deg_i, int* __restrict__ deg_o, int E) {
    int e = blockIdx.x * blockDim.x + threadIdx.x;
    if (e < E) {
        atomicAdd(&deg_o[dst[e]], 1);
        atomicAdd(&deg_i[src[e]], 1);
    }
}

// ---------------------------------------------------------------------------
// rel2[r] = edge_embs[r] @ W_rel^T + b_rel   (only R=200 unique rows!)
// ---------------------------------------------------------------------------
__global__ __launch_bounds__(DIM) void rel_transform_kernel(
    const float* __restrict__ edge_embs, const float* __restrict__ W_rel,
    const float* __restrict__ b_rel, float* __restrict__ rel2) {
    __shared__ float e[DIM];
    const int r = blockIdx.x;
    const int o = threadIdx.x;
    e[o] = edge_embs[(size_t)r * DIM + o];
    __syncthreads();
    float acc = b_rel[o];
    const float* w = W_rel + (size_t)o * DIM;
#pragma unroll 8
    for (int k = 0; k < DIM; ++k) acc = fmaf(e[k], w[k], acc);
    rel2[(size_t)r * DIM + o] = acc;
}

// ---------------------------------------------------------------------------
// Edge messages: one wave32 per edge, 4 feats/lane (float4).
//   ho_sum[dst] += h[src] - rel[evid]
//   hi_sum[src] += h[dst] - rel[evid]
// Gather tables are L2-resident (25.6 MB node feats, 100 KB rel table);
// the 8 fadd atomics/lane are the hot loop -> must be native f32 atomics.
// ---------------------------------------------------------------------------
__global__ __launch_bounds__(256) void edge_msg_kernel(
    const float* __restrict__ h, const float* __restrict__ rel,
    const int* __restrict__ src, const int* __restrict__ dst,
    const int* __restrict__ evid,
    float* __restrict__ ho_sum, float* __restrict__ hi_sum, int E) {
    const int wave = blockIdx.x * 8 + (threadIdx.x >> 5);
    const int lane = threadIdx.x & 31;
    if (wave >= E) return;
    const int s = src[wave], d = dst[wave], r = evid[wave];
    const int f = lane * 4;
    const float4 hs = *reinterpret_cast<const float4*>(h   + (size_t)s * DIM + f);
    const float4 hd = *reinterpret_cast<const float4*>(h   + (size_t)d * DIM + f);
    const float4 he = *reinterpret_cast<const float4*>(rel + (size_t)r * DIM + f);
    float* po = ho_sum + (size_t)d * DIM + f;
    float* pi = hi_sum + (size_t)s * DIM + f;
    atomic_add_f32(po + 0, hs.x - he.x);
    atomic_add_f32(po + 1, hs.y - he.y);
    atomic_add_f32(po + 2, hs.z - he.z);
    atomic_add_f32(po + 3, hs.w - he.w);
    atomic_add_f32(pi + 0, hd.x - he.x);
    atomic_add_f32(pi + 1, hd.y - he.y);
    atomic_add_f32(pi + 2, hd.z - he.z);
    atomic_add_f32(pi + 3, hd.w - he.w);
}

// ---------------------------------------------------------------------------
// WMMA fragment loaders (contiguous k = 16h + s order, same for A and B)
// A 16x32: lane (m = lane&15) holds 16 contiguous K at kb (lanes<16)
//          or kb+16 (lanes>=16), 2 bf16/VGPR.
// B 32x16: lane (n = lane&15) holds 16 contiguous K at kb (lanes<16)
//          or kb+16 (lanes>=16). Since B = W^T, that is W[n][kb..kb+15].
// ---------------------------------------------------------------------------
__device__ __forceinline__ v16bf load_a_frag(const float* __restrict__ X,
                                             int m0, int kb, float s, int lane) {
    const int m = m0 + (lane & 15);
    const float* p = X + (size_t)m * DIM + kb + ((lane & 16) ? 16 : 0);
    v16bf a;
#pragma unroll
    for (int i = 0; i < 8; ++i) a[i] = (__bf16)(p[i] * s);
#pragma unroll
    for (int i = 0; i < 8; ++i) a[8 + i] = (__bf16)(p[8 + i] * s);
    return a;
}

__device__ __forceinline__ v16bf load_b_frag(const float* __restrict__ W,
                                             int n0, int kb, int lane) {
    const int n = n0 + (lane & 15);
    const float* p = W + (size_t)n * DIM + kb + ((lane & 16) ? 16 : 0);
    v16bf b;
#pragma unroll
    for (int i = 0; i < 16; ++i) b[i] = (__bf16)p[i];
    return b;
}

// ---------------------------------------------------------------------------
// Fused 3-GEMM combine + BN-stats epilogue:
//   pre = (ho_mean@W_O^T + hi_mean@W_I^T + h@W_S^T + b_O+b_I+b_S) / 3
// One 16x16 output tile per wave; the 8 waves of a block tile all 128 output
// columns disjointly, so per-column sum/sumsq partials reduce with a single
// shfl_xor(16) and one atomic per (column, block) -- no separate stats pass.
// Requires N % 16 == 0 (N = 50000 -> 3125 row tiles).
// ---------------------------------------------------------------------------
__global__ __launch_bounds__(256) void combine_gemm_kernel(
    const float* __restrict__ ho_sum, const float* __restrict__ hi_sum,
    const float* __restrict__ h,
    const int* __restrict__ deg_o, const int* __restrict__ deg_i,
    const float* __restrict__ W_O, const float* __restrict__ W_I,
    const float* __restrict__ W_S,
    const float* __restrict__ b_O, const float* __restrict__ b_I,
    const float* __restrict__ b_S,
    float* __restrict__ pre,
    float* __restrict__ s_sum, float* __restrict__ s_sq) {
    const int lane = threadIdx.x & 31;
    const int wave = threadIdx.x >> 5;          // 0..7 -> output column tile
    const int m0 = blockIdx.x * 16;
    const int n0 = wave * 16;

    const int mrow = m0 + (lane & 15);
    const int co = deg_o[mrow];
    const int ci = deg_i[mrow];
    const float so = (co > 0) ? 1.0f / (float)co : 0.0f;   // zero rows stay zero
    const float si = (ci > 0) ? 1.0f / (float)ci : 0.0f;

    const float* Xs[3] = {ho_sum, hi_sum, h};
    const float* Ws[3] = {W_O, W_I, W_S};
    const float  sc[3] = {so, si, 1.0f};

    v8f acc = {};
#pragma unroll
    for (int j = 0; j < 3; ++j) {
#pragma unroll
        for (int kb = 0; kb < DIM; kb += 32) {
            v16bf a = load_a_frag(Xs[j], m0, kb, sc[j], lane);
            v16bf b = load_b_frag(Ws[j], n0, kb, lane);
            acc = __builtin_amdgcn_wmma_f32_16x16x32_bf16(
                false, a, false, b, (short)0, acc, false, false);
        }
    }

    const int n = n0 + (lane & 15);
    const float bias = b_O[n] + b_I[n] + b_S[n];
    const int mb = m0 + ((lane & 16) ? 8 : 0);
    float ps = 0.0f, pss = 0.0f;
#pragma unroll
    for (int r = 0; r < 8; ++r) {
        const float v = (acc[r] + bias) * (1.0f / 3.0f);
        pre[(size_t)(mb + r) * DIM + n] = v;
        ps += v;
        pss += v * v;
    }
    // lanes L and L+16 hold the same column (rows mb and mb+8): pair-reduce.
    ps  += __shfl_xor(ps, 16);
    pss += __shfl_xor(pss, 16);
    if (lane < 16) {
        atomic_add_f32(&s_sum[n], ps);
        atomic_add_f32(&s_sq[n], pss);
    }
}

// Fold mean/var/gamma/beta into per-column scale+shift (in place).
__global__ void bn_finalize_kernel(float* __restrict__ s_sum, float* __restrict__ s_sq,
                                   const float* __restrict__ g,
                                   const float* __restrict__ b, int N) {
    const int c = threadIdx.x;
    const float inv = 1.0f / (float)N;
    const float mu = s_sum[c] * inv;
    const float var = s_sq[c] * inv - mu * mu;   // biased var (training-mode BN)
    const float scale = g[c] * rsqrtf(var + BN_EPS);
    s_sum[c] = scale;
    s_sq[c] = b[c] - mu * scale;
}

__global__ void bn_apply_kernel(const float* __restrict__ pre,
                                const float* __restrict__ scale,
                                const float* __restrict__ shift,
                                float* __restrict__ out, size_t total, int do_tanh) {
    size_t i = (size_t)blockIdx.x * blockDim.x + threadIdx.x;
    size_t stride = (size_t)gridDim.x * blockDim.x;
    for (; i < total; i += stride) {
        const int c = (int)(i & (DIM - 1));
        const float v = pre[i] * scale[c] + shift[c];
        out[i] = do_tanh ? tanhf(v) : v;
    }
}

// ---------------------------------------------------------------------------
// Query vectors: q[b] = heads+rels (tail pred) or tails-rels (head pred),
// so dist[b][n] = sum_c |hn[n][c] - q[b][c]| in both cases.
// ---------------------------------------------------------------------------
__global__ void qvec_kernel(const float* __restrict__ hn, const float* __restrict__ rel2,
                            const int* __restrict__ hids, const int* __restrict__ rids,
                            const int* __restrict__ tids, const int* __restrict__ is_head,
                            float* __restrict__ q, int B) {
    const int i = blockIdx.x * blockDim.x + threadIdx.x;
    if (i >= B * DIM) return;
    const int b = i >> 7;
    const int c = i & (DIM - 1);
    const float rel = rel2[(size_t)rids[b] * DIM + c];
    float v;
    if (*is_head == 1)
        v = hn[(size_t)tids[b] * DIM + c] - rel;
    else
        v = hn[(size_t)hids[b] * DIM + c] + rel;
    q[i] = v;
}

// ---------------------------------------------------------------------------
// TransE scoring: score[b][n] = sigmoid(GAMMA - L1(hn[n], q[b])).
// Queries + 8-node tiles staged in LDS; thread = (local node, b).
// ---------------------------------------------------------------------------
__global__ __launch_bounds__(256) void score_kernel(
    const float* __restrict__ hn, const float* __restrict__ q,
    float* __restrict__ score, int N, int B) {
    __shared__ float qs[32 * DIM];
    __shared__ float hs[8][DIM];
    for (int i = threadIdx.x; i < B * DIM; i += blockDim.x) qs[i] = q[i];

    const int nl = threadIdx.x >> 5;     // 0..7 local node
    const int b = threadIdx.x & 31;      // query index (B == 32)
    const int base = blockIdx.x * 64;

    for (int chunk = 0; chunk < 64; chunk += 8) {
        const int n0 = base + chunk;
        __syncthreads();
        for (int i = threadIdx.x; i < 8 * DIM; i += blockDim.x) {
            const int rr = n0 + (i >> 7);
            hs[i >> 7][i & (DIM - 1)] = (rr < N) ? hn[(size_t)rr * DIM + (i & (DIM - 1))] : 0.0f;
        }
        __syncthreads();
        const int n = n0 + nl;
        if (n < N && b < B) {
            float dist = 0.0f;
#pragma unroll 8
            for (int c = 0; c < DIM; ++c)
                dist += fabsf(hs[nl][c] - qs[b * DIM + c]);
            score[(size_t)b * N + n] = 1.0f / (1.0f + __expf(dist - GAMMA));
        }
    }
}

// ---------------------------------------------------------------------------
// Host orchestration
// ---------------------------------------------------------------------------
extern "C" void kernel_launch(void* const* d_in, const int* in_sizes, int n_in,
                              void* d_out, int out_size, void* d_ws, size_t ws_size,
                              hipStream_t stream) {
    (void)n_in; (void)out_size; (void)ws_size;

    const float* node_embs = (const float*)d_in[0];
    const float* edge_embs = (const float*)d_in[1];
    const int*   src       = (const int*)d_in[2];
    const int*   dst       = (const int*)d_in[3];
    const int*   evid      = (const int*)d_in[4];
    const int*   hids      = (const int*)d_in[5];
    const int*   rids      = (const int*)d_in[6];
    const int*   tids      = (const int*)d_in[7];
    const float* W_O1 = (const float*)d_in[8];
    const float* b_O1 = (const float*)d_in[9];
    const float* W_I1 = (const float*)d_in[10];
    const float* b_I1 = (const float*)d_in[11];
    const float* W_S1 = (const float*)d_in[12];
    const float* b_S1 = (const float*)d_in[13];
    const float* W_rel1 = (const float*)d_in[14];
    const float* b_rel1 = (const float*)d_in[15];
    const float* W_O2 = (const float*)d_in[16];
    const float* b_O2 = (const float*)d_in[17];
    const float* W_I2 = (const float*)d_in[18];
    const float* b_I2 = (const float*)d_in[19];
    const float* W_S2 = (const float*)d_in[20];
    const float* b_S2 = (const float*)d_in[21];
    const float* bn_g1 = (const float*)d_in[22];
    const float* bn_b1 = (const float*)d_in[23];
    const float* bn_g2 = (const float*)d_in[24];
    const float* bn_b2 = (const float*)d_in[25];
    const int*   is_head = (const int*)d_in[26];

    const int N = in_sizes[0] / DIM;     // 50000
    const int R = in_sizes[1] / DIM;     // 200
    const int E = in_sizes[2];           // 640000
    const int B = in_sizes[5];           // 32
    const size_t nd = (size_t)N * DIM;

    // Workspace layout (~103 MB)
    char* ws = (char*)d_ws;
    float* ho_sum = (float*)ws;  ws += nd * sizeof(float);
    float* hi_sum = (float*)ws;  ws += nd * sizeof(float);   // contiguous with ho_sum
    float* pre    = (float*)ws;  ws += nd * sizeof(float);
    float* hn1    = (float*)ws;  ws += nd * sizeof(float);
    int*   deg_o  = (int*)ws;    ws += (size_t)N * sizeof(int);
    int*   deg_i  = (int*)ws;    ws += (size_t)N * sizeof(int);
    float* rel2   = (float*)ws;  ws += (size_t)R * DIM * sizeof(float);
    float* qv     = (float*)ws;  ws += (size_t)B * DIM * sizeof(float);
    float* s_sum  = (float*)ws;  ws += DIM * sizeof(float);
    float* s_sq   = (float*)ws;  ws += DIM * sizeof(float);  // contiguous with s_sum

    float* hn2   = (float*)d_out;            // [N, DIM]
    float* score = (float*)d_out + nd;       // [B, N]

    const int edge_blocks = (E + 7) / 8;     // one wave32 per edge, 8 waves/block
    const int row_tiles = N / 16;            // N % 16 == 0

    // Degrees (shared by both layers); int zeroing via float-bits zero is fine.
    zero_kernel<<<256, 256, 0, stream>>>((float*)deg_o, (size_t)N * 2);
    degree_kernel<<<(E + 255) / 256, 256, 0, stream>>>(src, dst, deg_i, deg_o, E);

    // Transform the 200-row relation table once (instead of a 640k-row GEMM)
    rel_transform_kernel<<<R, DIM, 0, stream>>>(edge_embs, W_rel1, b_rel1, rel2);

    // ---- Layer 1 ----
    zero_kernel<<<4096, 256, 0, stream>>>(ho_sum, nd * 2);      // ho_sum + hi_sum
    zero_kernel<<<1, 256, 0, stream>>>(s_sum, DIM * 2);         // s_sum + s_sq
    edge_msg_kernel<<<edge_blocks, 256, 0, stream>>>(node_embs, edge_embs, src, dst,
                                                     evid, ho_sum, hi_sum, E);
    combine_gemm_kernel<<<row_tiles, 256, 0, stream>>>(ho_sum, hi_sum, node_embs,
                                                       deg_o, deg_i,
                                                       W_O1, W_I1, W_S1,
                                                       b_O1, b_I1, b_S1,
                                                       pre, s_sum, s_sq);
    bn_finalize_kernel<<<1, DIM, 0, stream>>>(s_sum, s_sq, bn_g1, bn_b1, N);
    bn_apply_kernel<<<2048, 256, 0, stream>>>(pre, s_sum, s_sq, hn1, nd, /*tanh=*/1);

    // ---- Layer 2 ----
    zero_kernel<<<4096, 256, 0, stream>>>(ho_sum, nd * 2);
    zero_kernel<<<1, 256, 0, stream>>>(s_sum, DIM * 2);
    edge_msg_kernel<<<edge_blocks, 256, 0, stream>>>(hn1, rel2, src, dst,
                                                     evid, ho_sum, hi_sum, E);
    combine_gemm_kernel<<<row_tiles, 256, 0, stream>>>(ho_sum, hi_sum, hn1,
                                                       deg_o, deg_i,
                                                       W_O2, W_I2, W_S2,
                                                       b_O2, b_I2, b_S2,
                                                       pre, s_sum, s_sq);
    bn_finalize_kernel<<<1, DIM, 0, stream>>>(s_sum, s_sq, bn_g2, bn_b2, N);
    bn_apply_kernel<<<2048, 256, 0, stream>>>(pre, s_sum, s_sq, hn2, nd, /*tanh=*/0);

    // ---- Scoring ----
    qvec_kernel<<<(B * DIM + 255) / 256, 256, 0, stream>>>(hn2, rel2, hids, rids,
                                                           tids, is_head, qv, B);
    score_kernel<<<(N + 63) / 64, 256, 0, stream>>>(hn2, qv, score, N, B);
}
